// GEN_NET_56891136803144
// MI455X (gfx1250) — matrix operand order repaired and checked by reading, not compiled, not run on hardware
//
#include <hip/hip_runtime.h>

typedef __attribute__((ext_vector_type(16))) _Float16 v16h;
typedef __attribute__((ext_vector_type(8)))  float    v8f;

#define NNODES 65536
#define NEDGES 1048576

// ---------------------------------------------------------------- utilities
__global__ __launch_bounds__(256) void zero_f32(float* __restrict__ p, int n) {
  int i = blockIdx.x * blockDim.x + threadIdx.x;
  if (i < n) p[i] = 0.0f;
}

// ------------------------------------------------- B-fragment pre-packing
// Packs W[K,N] (fp32, row-major) into per-lane WMMA B fragments (f16):
//   Wp[((chunk*nTiles + nT)*32 + lane)] = v16h with halves h=0..15 holding
//   W[(chunk*32 + (lane>=16?8:0) + (h&7) + (h>=8?16:0)) * N + nT*16 + lane%16]   (0 if k>=K)
__global__ __launch_bounds__(256)
void pack_w(const float* __restrict__ W, v16h* __restrict__ Wp, int K, int N) {
  int idx = blockIdx.x * blockDim.x + threadIdx.x;
  const int nTiles = N >> 4;
  const int KC = (K + 31) >> 5;
  if (idx >= KC * nTiles * 32) return;
  const int lane = idx & 31;
  const int nT = (idx >> 5) % nTiles;
  const int c  = (idx >> 5) / nTiles;
  const int col = (nT << 4) + (lane & 15);
  const int kbase = (c << 5) + ((lane >> 4) << 3);
  v16h b;
#pragma unroll
  for (int h = 0; h < 16; ++h) {
    const int k = kbase + (h & 7) + ((h >> 3) << 4);
    b[h] = (_Float16)((k < K) ? W[k * N + col] : 0.0f);
  }
  Wp[idx] = b;
}

// ---------------------------------------------------------------- WMMA GEMM
// C[M,N] = epilogue(A[M,K] @ W[K,N] + bias); one wave32 per 16xN STRIP:
// the A fragment is loaded once per 32-K chunk and reused across all N tiles
// (weights are tiny and L2-resident), minimizing A traffic.
// A fragment (16-bit 16x32, ISA 7.12.2) = two contiguous 8-float runs/lane:
//   halves 0-7  -> K = kc + hi*8 + h ;  halves 8-15 -> K = kc + 16 + hi*8 + h
// so A loads are 4x float4 (global_load_b128). B comes pre-packed (1x v16h).
enum { EPI_BIAS = 0, EPI_BN_RELU = 1, EPI_RELU = 2 };

template <int K, int N, int EPI>
__global__ __launch_bounds__(256)
void gemm_wmma_f16(const float* __restrict__ A, const v16h* __restrict__ Wp,
                   const float* __restrict__ bias, const float* __restrict__ g,
                   const float* __restrict__ be, float* __restrict__ C) {
  constexpr int nTiles = N >> 4;
  constexpr int KC = (K + 31) >> 5;
  const int lane = threadIdx.x & 31;
  const int wave = threadIdx.x >> 5;
  const int mT = blockIdx.x * (blockDim.x >> 5) + wave;
  if (mT >= (NNODES >> 4)) return;             // uniform per wave
  const int hi   = lane >> 4;
  const int row  = (mT << 4) + (lane & 15);
  const int ncol = lane & 15;
  const float* __restrict__ arow = A + (size_t)row * K;

  v8f acc[nTiles];
#pragma unroll
  for (int nT = 0; nT < nTiles; ++nT) acc[nT] = (v8f){};

#pragma unroll
  for (int c = 0; c < KC; ++c) {
    const int kc = c << 5;
    float f[16];
    *(float4*)(f + 0) = *(const float4*)(arow + kc + hi * 8);
    *(float4*)(f + 4) = *(const float4*)(arow + kc + hi * 8 + 4);
    if constexpr (K % 32 == 0) {
      *(float4*)(f + 8)  = *(const float4*)(arow + kc + 16 + hi * 8);
      *(float4*)(f + 12) = *(const float4*)(arow + kc + 16 + hi * 8 + 4);
    } else {
#pragma unroll
      for (int h = 8; h < 16; ++h) f[h] = 0.0f;   // K == 16 zero-pad
    }
    v16h a;
#pragma unroll
    for (int h = 0; h < 16; ++h) a[h] = (_Float16)f[h];
#pragma unroll
    for (int nT = 0; nT < nTiles; ++nT) {
      const v16h b = Wp[(c * nTiles + nT) * 32 + lane];
      acc[nT] = __builtin_amdgcn_wmma_f32_16x16x32_f16(false, a, false, b,
                                                       (short)0, acc[nT],
                                                       false, false);
    }
  }

  const float bnrs = 0.9999950000374997f;      // 1/sqrt(1 + 1e-5)
#pragma unroll
  for (int nT = 0; nT < nTiles; ++nT) {
    const int col = (nT << 4) + ncol;
    const float bv = bias[col];
    float scale = 0.0f, shift = 0.0f;
    if (EPI == EPI_BN_RELU) { scale = g[col] * bnrs; shift = be[col]; }
#pragma unroll
    for (int r = 0; r < 8; ++r) {
      const int m = (mT << 4) + r + (hi << 3);
      float val = acc[nT][r] + bv;
      if (EPI == EPI_BN_RELU) val = fmaxf(val * scale + shift, 0.0f);
      else if (EPI == EPI_RELU) val = fmaxf(val, 0.0f);
      C[(size_t)m * N + col] = val;
    }
  }
}

// ------------------------------------------------------- edge pass 1: max
// msg = relu(h[src]) + 1e-7 > 0, so bit-pattern uint atomicMax is exact and a
// zero-initialized buffer reproduces the reference's "isolated node -> 0".
template <int D>
__global__ __launch_bounds__(256)
void edge_max(const float* __restrict__ h, const long long* __restrict__ src,
              const long long* __restrict__ dst, float* __restrict__ mx) {
  int e = blockIdx.x * blockDim.x + threadIdx.x;
  if (e >= NEDGES) return;
  const long long s = src[e], d = dst[e];
  const float4* hr = (const float4*)(h + s * D);
  unsigned int* mr = (unsigned int*)(mx + d * D);
#pragma unroll 4
  for (int c = 0; c < D / 4; ++c) {
    float4 hv = hr[c];
    atomicMax(&mr[4 * c + 0], __float_as_uint(fmaxf(hv.x, 0.0f) + 1e-7f));
    atomicMax(&mr[4 * c + 1], __float_as_uint(fmaxf(hv.y, 0.0f) + 1e-7f));
    atomicMax(&mr[4 * c + 2], __float_as_uint(fmaxf(hv.z, 0.0f) + 1e-7f));
    atomicMax(&mr[4 * c + 3], __float_as_uint(fmaxf(hv.w, 0.0f) + 1e-7f));
  }
}

// -------------------------------------- edge pass 2: denominator + numerator
// softmax aggr folds to (sum e*msg)/(sum e + eps) -> only two edge passes.
template <int D>
__global__ __launch_bounds__(256)
void edge_accum(const float* __restrict__ h, const long long* __restrict__ src,
                const long long* __restrict__ dst, const float* __restrict__ mx,
                float* __restrict__ ssum, float* __restrict__ num) {
  int e = blockIdx.x * blockDim.x + threadIdx.x;
  if (e >= NEDGES) return;
  const long long s = src[e], d = dst[e];
  const float4* hr = (const float4*)(h + s * D);
  const float4* mr = (const float4*)(mx + d * D);
  float* sr = ssum + d * D;
  float* nr = num + d * D;
#pragma unroll 4
  for (int c = 0; c < D / 4; ++c) {
    float4 hv = hr[c];
    float4 mv = mr[c];
    float m0 = fmaxf(hv.x, 0.0f) + 1e-7f, e0 = __expf(m0 - mv.x);
    float m1 = fmaxf(hv.y, 0.0f) + 1e-7f, e1 = __expf(m1 - mv.y);
    float m2 = fmaxf(hv.z, 0.0f) + 1e-7f, e2 = __expf(m2 - mv.z);
    float m3 = fmaxf(hv.w, 0.0f) + 1e-7f, e3 = __expf(m3 - mv.w);
    unsafeAtomicAdd(&sr[4 * c + 0], e0);
    unsafeAtomicAdd(&sr[4 * c + 1], e1);
    unsafeAtomicAdd(&sr[4 * c + 2], e2);
    unsafeAtomicAdd(&sr[4 * c + 3], e3);
    unsafeAtomicAdd(&nr[4 * c + 0], e0 * m0);
    unsafeAtomicAdd(&nr[4 * c + 1], e1 * m1);
    unsafeAtomicAdd(&nr[4 * c + 2], e2 * m2);
    unsafeAtomicAdd(&nr[4 * c + 3], e3 * m3);
  }
}

// ------------------------------------------- residual + softmax aggregation
template <int D>
__global__ __launch_bounds__(256)
void finalize_agg(float* __restrict__ h, const float* __restrict__ num,
                  const float* __restrict__ ssum) {
  int i = blockIdx.x * blockDim.x + threadIdx.x;
  if (i < NNODES * D) h[i] = h[i] + num[i] / (ssum[i] + 1e-16f);
}

// ----------------------------------------------------- row-wise log_softmax
__global__ __launch_bounds__(256) void log_softmax16(float* __restrict__ o) {
  int n = blockIdx.x * blockDim.x + threadIdx.x;
  if (n >= NNODES) return;
  float4* r4 = (float4*)(o + n * 16);
  float v[16];
#pragma unroll
  for (int c = 0; c < 4; ++c) *(float4*)(v + 4 * c) = r4[c];
  float m = v[0];
#pragma unroll
  for (int c = 1; c < 16; ++c) m = fmaxf(m, v[c]);
  float sum = 0.0f;
#pragma unroll
  for (int c = 0; c < 16; ++c) sum += __expf(v[c] - m);
  const float lse = m + __logf(sum);
#pragma unroll
  for (int c = 0; c < 16; ++c) v[c] -= lse;
#pragma unroll
  for (int c = 0; c < 4; ++c) r4[c] = *(float4*)(v + 4 * c);
}

// ---------------------------------------------------------------- launcher
extern "C" void kernel_launch(void* const* d_in, const int* in_sizes, int n_in,
                              void* d_out, int out_size, void* d_ws, size_t ws_size,
                              hipStream_t stream) {
  const float* x      = (const float*)d_in[0];
  const long long* ei = (const long long*)d_in[1];   // int64 per reference
  const float* lin1_W = (const float*)d_in[2];
  const float* lin1_b = (const float*)d_in[3];
  const float* W1_1   = (const float*)d_in[4];
  const float* b1_1   = (const float*)d_in[5];
  const float* g1     = (const float*)d_in[6];
  const float* be1    = (const float*)d_in[7];
  const float* W2_1   = (const float*)d_in[8];
  const float* b2_1   = (const float*)d_in[9];
  const float* lin2_W = (const float*)d_in[10];
  const float* lin2_b = (const float*)d_in[11];
  const float* W1_2   = (const float*)d_in[12];
  const float* b1_2   = (const float*)d_in[13];
  const float* g2     = (const float*)d_in[14];
  const float* be2    = (const float*)d_in[15];
  const float* W2_2   = (const float*)d_in[16];
  const float* b2_2   = (const float*)d_in[17];
  float* out = (float*)d_out;

  const long long* src = ei;            // edge_index[0, :]
  const long long* dst = ei + NEDGES;   // edge_index[1, :]

  // workspace: 64 MB activations (reused) + packed weight fragments
  float* WS  = (float*)d_ws;
  float* h1  = WS;                 // 65536*64
  float* mx1 = WS + 4194304;       // 65536*64 (later: r)
  float* s1  = WS + 8388608;       // 65536*64 (later: z1 / z2)
  float* nm1 = WS + 12582912;      // 65536*64 (later: z1 upper half)
  v16h* WP   = (v16h*)(WS + 16777216);
  v16h* WpL1 = WP;                 // K=64  N=64  -> 2*4*32  = 256 frags
  v16h* Wp11 = WP + 256;           // K=64  N=128 -> 2*8*32  = 512
  v16h* Wp21 = WP + 768;           // K=128 N=64  -> 4*4*32  = 512
  v16h* WpL2 = WP + 1280;          // K=64  N=16  -> 2*1*32  = 64
  v16h* Wp12 = WP + 1344;          // K=16  N=32  -> 1*2*32  = 64
  v16h* Wp22 = WP + 1408;          // K=32  N=16  -> 1*1*32  = 32

  dim3 b256(256);
  dim3 gE(NEDGES / 256);
  dim3 gStrip(512);                // 4096 row-strips / 8 waves per block

  // pack all weight matrices into WMMA B fragments
  pack_w<<<dim3(1), b256, 0, stream>>>(lin1_W, WpL1, 64, 64);
  pack_w<<<dim3(2), b256, 0, stream>>>(W1_1,   Wp11, 64, 128);
  pack_w<<<dim3(2), b256, 0, stream>>>(W2_1,   Wp21, 128, 64);
  pack_w<<<dim3(1), b256, 0, stream>>>(lin2_W, WpL2, 64, 16);
  pack_w<<<dim3(1), b256, 0, stream>>>(W1_2,   Wp12, 16, 32);
  pack_w<<<dim3(1), b256, 0, stream>>>(W2_2,   Wp22, 32, 16);

  // ---------------- layer 1 (D = 64) ----------------
  zero_f32<<<dim3((3 * 4194304) / 256), b256, 0, stream>>>(mx1, 3 * 4194304);
  gemm_wmma_f16<64, 64, EPI_BIAS><<<gStrip, b256, 0, stream>>>(
      x, WpL1, lin1_b, nullptr, nullptr, h1);
  edge_max<64><<<gE, b256, 0, stream>>>(h1, src, dst, mx1);
  edge_accum<64><<<gE, b256, 0, stream>>>(h1, src, dst, mx1, s1, nm1);
  finalize_agg<64><<<dim3((NNODES * 64) / 256), b256, 0, stream>>>(h1, nm1, s1);
  float* z1 = s1;                  // 65536*128 overlays s1+nm1 (dead)
  gemm_wmma_f16<64, 128, EPI_BN_RELU><<<gStrip, b256, 0, stream>>>(
      h1, Wp11, b1_1, g1, be1, z1);
  float* r = mx1;                  // 65536*64 overlays mx1 (dead)
  gemm_wmma_f16<128, 64, EPI_RELU><<<gStrip, b256, 0, stream>>>(
      z1, Wp21, b2_1, nullptr, nullptr, r);

  // ---------------- layer 2 (D = 16) ----------------
  float* h2  = WS;                 // overlays h1 slab (dead)
  float* mx2 = WS + 1048576;
  float* s2  = WS + 2097152;
  float* nm2 = WS + 3145728;
  zero_f32<<<dim3((3 * 1048576) / 256), b256, 0, stream>>>(mx2, 3 * 1048576);
  gemm_wmma_f16<64, 16, EPI_BIAS><<<gStrip, b256, 0, stream>>>(
      r, WpL2, lin2_b, nullptr, nullptr, h2);
  edge_max<16><<<gE, b256, 0, stream>>>(h2, src, dst, mx2);
  edge_accum<16><<<gE, b256, 0, stream>>>(h2, src, dst, mx2, s2, nm2);
  finalize_agg<16><<<dim3((NNODES * 16) / 256), b256, 0, stream>>>(h2, nm2, s2);
  float* z2 = s1;                  // 65536*32 overlays z1 slab (dead)
  gemm_wmma_f16<16, 32, EPI_BN_RELU><<<gStrip, b256, 0, stream>>>(
      h2, Wp12, b1_2, g2, be2, z2);
  gemm_wmma_f16<32, 16, EPI_BIAS><<<gStrip, b256, 0, stream>>>(
      z2, Wp22, b2_2, nullptr, nullptr, out);
  log_softmax16<<<dim3(NNODES / 256), b256, 0, stream>>>(out);
}
